// GraphConv_360777253122
// MI455X (gfx1250) — matrix operand order repaired and checked by reading, not compiled, not run on hardware
//
#include <hip/hip_runtime.h>
#include <hip/hip_bf16.h>

#define HCH 128   // hidden/in channels (fixed by reference)

typedef __attribute__((ext_vector_type(16))) __bf16 v16bf;
typedef __attribute__((ext_vector_type(8)))  float  v8f;

static __device__ __forceinline__ __bf16 f2bf(float f) {
    union { float f; unsigned u; } a; a.f = f;
    unsigned r = a.u + 0x7FFFu + ((a.u >> 16) & 1u);   // round-to-nearest-even
    unsigned short h = (unsigned short)(r >> 16);
    return __builtin_bit_cast(__bf16, h);
}

// ---------------- degree histogram -------------------------------------------
__global__ void deg_kernel(const int* __restrict__ col, float* __restrict__ deg, int E) {
    int e = blockIdx.x * blockDim.x + threadIdx.x;
    if (e < E) atomicAdd(&deg[col[e]], 1.0f);
}

// ---------------- WMMA GEMM: C[nrows x 128] = A[nrows x 128] * W[128 x 128] + b
// Block: 256 thr / 8 waves. Each wave owns a 16x32 tile (2 accumulators, shared
// A fragment). Block covers a 32-row slab; grid-stride over slabs amortizes the
// one-time fp32->bf16 staging of W (transposed) in LDS.
__global__ __launch_bounds__(256)
void gemm_bias_kernel(const float* __restrict__ A, const float* __restrict__ W,
                      const float* __restrict__ bias, float* __restrict__ C,
                      int nrows, int nslabs)
{
    __shared__ __bf16 ldsW[HCH][HCH + 8];   // transposed: ldsW[n][k] = W[k][n]
    __shared__ __bf16 ldsA[32][HCH + 8];    // ldsA[m][k]

    const int tid  = threadIdx.x;
    const int wave = tid >> 5;
    const int lane = tid & 31;

    // stage W (fp32 -> bf16, transposed) once per block
    for (int idx = tid; idx < HCH * HCH; idx += 256) {
        int k = idx >> 7, n = idx & 127;
        ldsW[n][k] = f2bf(W[idx]);
    }

    const int half  = lane >> 4;            // 0: lanes 0-15, 1: lanes 16-31
    const int mn    = lane & 15;            // M for A-frag, N for B-frag / C
    const int rquad = wave >> 2;            // 16-row sub-slab (0/1)
    const int colBase = (wave & 3) * 32;    // 4 col groups of 32
    const float bv0 = bias[colBase + mn];
    const float bv1 = bias[colBase + 16 + mn];

    for (int slab = blockIdx.x; slab < nslabs; slab += gridDim.x) {
        const int rowBase = slab * 32;
        __syncthreads();                    // prior readers done before re-stage
        for (int idx = tid; idx < 32 * HCH; idx += 256) {
            int m = idx >> 7, k = idx & 127;
            int r = rowBase + m;
            ldsA[m][k] = f2bf(r < nrows ? A[(long)r * HCH + k] : 0.0f);
        }
        __syncthreads();

        v8f acc0 = {}, acc1 = {};
        for (int ks = 0; ks < HCH; ks += 32) {
            // A 16x32 bf16 frag: lane-half K range [half*16, +16), contiguous
            v16bf af, b0, b1;
            const int ab = ks + half * 16;
            const int am = rquad * 16 + mn;
#pragma unroll
            for (int i = 0; i < 8; ++i) {
                af[i]     = ldsA[am][ab + i];
                af[8 + i] = ldsA[am][ab + 8 + i];
            }
            // B 32x16 bf16 frag: lane-half K range [half*16, +16), contiguous
            const int bb = ks + half * 16;
#pragma unroll
            for (int i = 0; i < 16; ++i) {
                b0[i] = ldsW[colBase + mn][bb + i];
                b1[i] = ldsW[colBase + 16 + mn][bb + i];
            }
            acc0 = __builtin_amdgcn_wmma_f32_16x16x32_bf16(
                       false, af, false, b0, (short)0, acc0, false, false);
            acc1 = __builtin_amdgcn_wmma_f32_16x16x32_bf16(
                       false, af, false, b1, (short)0, acc1, false, false);
        }

        // C/D layout: VGPR r -> M = r + half*8 (within the 16-row tile)
        const long mBase = (long)rowBase + rquad * 16 + half * 8;
        const long cb    = colBase + mn;
        if (rowBase + 32 <= nrows) {        // uniform fast path (no tail)
#pragma unroll
            for (int r = 0; r < 8; ++r) {
                long m = mBase + r;
                C[m * HCH + cb]      = acc0[r] + bv0;
                C[m * HCH + cb + 16] = acc1[r] + bv1;
            }
        } else {
#pragma unroll
            for (int r = 0; r < 8; ++r) {
                long m = mBase + r;
                if (m < nrows) {
                    C[m * HCH + cb]      = acc0[r] + bv0;
                    C[m * HCH + cb + 16] = acc1[r] + bv1;
                }
            }
        }
    }
}

// ---------------- SpMM: agg[col] += val(e) * h[row] (float4 per thread) ------
__global__ void spmm_kernel(const float* __restrict__ h, const int* __restrict__ row,
                            const int* __restrict__ col, const float* __restrict__ ew,
                            const float* __restrict__ deg, float* __restrict__ agg,
                            long totalQuads)
{
    long idx = (long)blockIdx.x * blockDim.x + threadIdx.x;
    if (idx >= totalQuads) return;
    int e  = (int)(idx >> 5);               // 32 quads per edge
    int c4 = (int)(idx & 31) << 2;
    int r = row[e], d = col[e];
    float dc = deg[d], dr = deg[r];
    float v = (dc > 0.0f && dr > 0.0f) ? ew[e] * rsqrtf(dc) * rsqrtf(dr) : 0.0f;
    if (v != 0.0f) {
        const float4 hv = *(const float4*)(h + (long)r * HCH + c4);
        float* ap = agg + (long)d * HCH + c4;
        atomicAdd(ap + 0, v * hv.x);
        atomicAdd(ap + 1, v * hv.y);
        atomicAdd(ap + 2, v * hv.z);
        atomicAdd(ap + 3, v * hv.w);
    }
}

// ---------------- BatchNorm: stats -> finalize -> apply ----------------------
__global__ __launch_bounds__(128)
void bn_stats_kernel(const float* __restrict__ pre, float* __restrict__ stats,
                     int nrows, int rowsPerBlock)
{
    int c  = threadIdx.x;
    int r0 = blockIdx.x * rowsPerBlock;
    int r1 = r0 + rowsPerBlock; if (r1 > nrows) r1 = nrows;
    float s = 0.0f, q = 0.0f;
    for (int r = r0; r < r1; ++r) {
        float v = pre[(long)r * HCH + c];
        s += v; q += v * v;
    }
    atomicAdd(&stats[c], s);
    atomicAdd(&stats[HCH + c], q);
}

__global__ void bn_finalize_kernel(float* __restrict__ stats,
                                   const float* __restrict__ gamma,
                                   const float* __restrict__ beta, float nInv)
{
    int c = threadIdx.x;
    float mean = stats[c] * nInv;
    float var  = stats[HCH + c] * nInv - mean * mean;
    var = fmaxf(var, 0.0f);
    float scale = gamma[c] * rsqrtf(var + 1e-5f);
    stats[c]       = scale;
    stats[HCH + c] = beta[c] - mean * scale;       // shift
}

__global__ void bn_apply_kernel(const float4* __restrict__ pre,
                                const float* __restrict__ stats,
                                const float4* __restrict__ res,
                                float4* __restrict__ out, long totalQuads)
{
    long idx = (long)blockIdx.x * blockDim.x + threadIdx.x;
    if (idx >= totalQuads) return;
    int c4 = (int)(idx & 31) << 2;                 // 32 quads per row
    float4 p = pre[idx];
    float4 o;
    o.x = fmaxf(fmaf(p.x, stats[c4 + 0], stats[HCH + c4 + 0]), 0.0f);
    o.y = fmaxf(fmaf(p.y, stats[c4 + 1], stats[HCH + c4 + 1]), 0.0f);
    o.z = fmaxf(fmaf(p.z, stats[c4 + 2], stats[HCH + c4 + 2]), 0.0f);
    o.w = fmaxf(fmaf(p.w, stats[c4 + 3], stats[HCH + c4 + 3]), 0.0f);
    if (res) {                                     // residual AFTER relu
        float4 rv = res[idx];
        o.x += rv.x; o.y += rv.y; o.z += rv.z; o.w += rv.w;
    }
    out[idx] = o;
}

// -----------------------------------------------------------------------------
extern "C" void kernel_launch(void* const* d_in, const int* in_sizes, int n_in,
                              void* d_out, int out_size, void* d_ws, size_t ws_size,
                              hipStream_t stream)
{
    const float* x      = (const float*)d_in[0];
    const int*   eidx   = (const int*)  d_in[1];
    const float* ew     = (const float*)d_in[2];
    const float* fc_w   = (const float*)d_in[3];
    const float* fc_b   = (const float*)d_in[4];
    const float* conv_w = (const float*)d_in[5];
    const float* conv_b = (const float*)d_in[6];
    const float* gamma  = (const float*)d_in[7];
    const float* beta   = (const float*)d_in[8];

    const int N = in_sizes[0] / HCH;     // 100000
    const int E = in_sizes[2];           // 1600000
    const int* rowp = eidx;              // edge_index[0]
    const int* colp = eidx + E;          // edge_index[1]

    // workspace carving
    float* deg   = (float*)d_ws;               // N
    float* stats = deg + N;                    // 256 (sum,sumsq)->(scale,shift)
    float* h     = stats + 256;                // N*128
    float* pre   = h + (long)N * HCH;          // N*128
    float* agg   = pre + (long)N * HCH;        // N*128

    const long  tot      = (long)N * HCH;
    const long  eQuads   = (long)E * (HCH / 4);
    const long  aQuads   = tot / 4;
    const int   ewBlocks = (int)((eQuads + 255) / 256);
    const int   apBlocks = (int)((aQuads + 255) / 256);
    const int   nslabs   = (N + 31) / 32;
    const int   gemmGrid = nslabs < 1024 ? nslabs : 1024;
    const int   rowsPerB = 256;
    const int   statGrid = (N + rowsPerB - 1) / rowsPerB;
    const float nInv     = 1.0f / (float)N;

    // degrees
    hipMemsetAsync(deg, 0, (size_t)N * sizeof(float), stream);
    deg_kernel<<<(E + 255) / 256, 256, 0, stream>>>(colp, deg, E);

    // input projection: pre = x @ fc_w + fc_b ; h = relu(bn0(pre))
    gemm_bias_kernel<<<gemmGrid, 256, 0, stream>>>(x, fc_w, fc_b, pre, N, nslabs);
    hipMemsetAsync(stats, 0, 256 * sizeof(float), stream);
    bn_stats_kernel<<<statGrid, 128, 0, stream>>>(pre, stats, N, rowsPerB);
    bn_finalize_kernel<<<1, 128, 0, stream>>>(stats, gamma, beta, nInv);
    bn_apply_kernel<<<apBlocks, 256, 0, stream>>>((const float4*)pre, stats,
                                                  nullptr, (float4*)h, aQuads);

    for (int l = 0; l < 2; ++l) {
        // aggregation
        hipMemsetAsync(agg, 0, (size_t)tot * sizeof(float), stream);
        spmm_kernel<<<ewBlocks, 256, 0, stream>>>(h, rowp, colp, ew, deg, agg,
                                                  eQuads);
        // pre = agg @ conv_w[l] + conv_b[l]
        gemm_bias_kernel<<<gemmGrid, 256, 0, stream>>>(
            agg, conv_w + (long)l * HCH * HCH, conv_b + (long)l * HCH, pre, N,
            nslabs);
        // bn(l+1) + relu + residual
        hipMemsetAsync(stats, 0, 256 * sizeof(float), stream);
        bn_stats_kernel<<<statGrid, 128, 0, stream>>>(pre, stats, N, rowsPerB);
        bn_finalize_kernel<<<1, 128, 0, stream>>>(stats,
                                                  gamma + (long)(l + 1) * HCH,
                                                  beta  + (long)(l + 1) * HCH, nInv);
        float* dst = (l == 1) ? (float*)d_out : h;
        bn_apply_kernel<<<apBlocks, 256, 0, stream>>>((const float4*)pre, stats,
                                                      (const float4*)h,
                                                      (float4*)dst, aQuads);
    }
}
